// CrossAttentionWithPhysicsBias_39487929319881
// MI455X (gfx1250) — compile-verified, matrix operand order checked by reading, not run on hardware
//
#include <hip/hip_runtime.h>
#include <hip/hip_bf16.h>
#include <stdint.h>

// CDNA5 / gfx1250: wave32, WMMA 16x16x32 bf16 (f32 accumulate).
typedef __attribute__((ext_vector_type(16))) __bf16 v16bf;
typedef __attribute__((ext_vector_type(8)))  __bf16 v8bf;
typedef __attribute__((ext_vector_type(8)))  float  v8f;

#define DMODEL 1024
#define NRES   256
#define NHEAD  16
#define NB     512
#define KTILE  32              // keys per streaming tile
#define NKT    (NRES / KTILE)  // 8

// ---------------------------------------------------------------------------
// WMMA fragment builders (per CDNA5 ISA 16-bit A/B VGPR layouts):
//  A[16x32] (m,k): lane r=L%16 -> row m=r; hi=L/16; elements e:
//     e<8  -> k = hi*8 + e        (contiguous 8 elements)
//     e>=8 -> k = 16 + hi*8 + e-8 (contiguous 8 elements)
//  B[32x16] (k,n) stored as [n][k] uses the identical pattern with row=n.
//  C[16x16]: element i -> m = i + hi*8, column n = r.
// ---------------------------------------------------------------------------
__device__ __forceinline__ v16bf frag_rowk(const __bf16* base, int stride,
                                           int row, int kb, int hi) {
  v16bf f;
  const __bf16* p = base + row * stride + kb + hi * 8;
#pragma unroll
  for (int e = 0; e < 8; ++e) { f[e] = p[e]; f[8 + e] = p[16 + e]; }
  return f;
}

// B fragment gathered from a [k][n] row-major buffer (column fixed per lane).
__device__ __forceinline__ v16bf frag_colgather(const __bf16* base, int stride,
                                                int col, int hi) {
  v16bf f;
#pragma unroll
  for (int e = 0; e < 8; ++e) {
    f[e]     = base[(hi * 8 + e) * stride + col];
    f[8 + e] = base[(16 + hi * 8 + e) * stride + col];
  }
  return f;
}

__device__ __forceinline__ v8bf pack8(float4 a, float4 b) {
  v8bf r;
  r[0] = (__bf16)a.x; r[1] = (__bf16)a.y;
  r[2] = (__bf16)a.z; r[3] = (__bf16)a.w;
  r[4] = (__bf16)b.x; r[5] = (__bf16)b.y;
  r[6] = (__bf16)b.z; r[7] = (__bf16)b.w;
  return r;
}

// async global->LDS DMA of one 16-byte packet per lane (ASYNCcnt tracked)
__device__ __forceinline__ void async_b128(uint32_t lds_ofs, const void* gsrc) {
  asm volatile("global_load_async_to_lds_b128 %0, %1, off"
               :: "v"(lds_ofs), "v"((unsigned long long)(uintptr_t)gsrc)
               : "memory");
}
__device__ __forceinline__ void async_drain() {
  asm volatile("s_wait_asynccnt 0" ::: "memory");
}

// ---------------------------------------------------------------------------
// Kernel 1: qs[b,c] = scale * (query[b] @ Wq^T + bq)[c]   (bf16 out)
// WMMA GEMM, M=512 (b), N=1024 (c), K=1024 (t). Block = 4 waves = 16b x 64c.
// ---------------------------------------------------------------------------
__global__ __launch_bounds__(128) void qproj_kernel(
    const float* __restrict__ query, const float* __restrict__ Wq,
    const float* __restrict__ bq, __bf16* __restrict__ qs) {
  __shared__ alignas(16) __bf16 q_lds[16][64];
  __shared__ alignas(16) __bf16 w_lds[64][64];   // [n=c][k=t]
  const int tid = threadIdx.x, wid = tid >> 5, lane = tid & 31;
  const int r = lane & 15, hi = lane >> 4;
  const int c0 = blockIdx.x * 64, b0 = blockIdx.y * 16;

  v8f acc;
#pragma unroll
  for (int i = 0; i < 8; ++i) acc[i] = 0.0f;

  for (int ch = 0; ch < 16; ++ch) {            // 16 K-chunks of 64
    const int t0 = ch * 64;
    {  // q tile: 128 groups of 8; 1 per thread
      int bl = tid >> 3, tl8 = (tid & 7) << 3;
      const float* src = query + (size_t)(b0 + bl) * DMODEL + t0 + tl8;
      *(v8bf*)&q_lds[bl][tl8] =
          pack8(*(const float4*)src, *(const float4*)(src + 4));
    }
#pragma unroll
    for (int it = 0; it < 4; ++it) {  // w tile: 512 groups of 8; 4 per thread
      int g = tid + it * 128;
      int cl = g >> 3, tl8 = (g & 7) << 3;
      const float* src = Wq + (size_t)(c0 + cl) * DMODEL + t0 + tl8;
      *(v8bf*)&w_lds[cl][tl8] =
          pack8(*(const float4*)src, *(const float4*)(src + 4));
    }
    __syncthreads();
#pragma unroll
    for (int kb = 0; kb < 64; kb += 32) {
      v16bf a = frag_rowk(&q_lds[0][0], 64, r, kb, hi);
      v16bf b = frag_rowk(&w_lds[0][0], 64, wid * 16 + r, kb, hi);
      acc = __builtin_amdgcn_wmma_f32_16x16x32_bf16(false, a, false, b,
                                                    (short)0, acc, false, false);
    }
    __syncthreads();
  }
  const float scale = 0.125f;  // DK^-0.5, folded into Q
#pragma unroll
  for (int i = 0; i < 8; ++i) {
    int bb = b0 + hi * 8 + i;
    int cc = c0 + wid * 16 + r;
    qs[(size_t)bb * DMODEL + cc] = (__bf16)(scale * (acc[i] + bq[cc]));
  }
}

// ---------------------------------------------------------------------------
// Kernel 2: A[b,h,j] = sum_d qs[b, h*64+d] * Wk[h*64+d, j]   (bf16 out)
// 16 per-head GEMMs, M=512 (b), N=1024 (j), K=64 (d). Note: bk adds a
// per-(b,h) constant to all keys' scores -> cancels in softmax, dropped.
// ---------------------------------------------------------------------------
__global__ __launch_bounds__(128) void akern(
    const __bf16* __restrict__ qs, const float* __restrict__ Wk,
    __bf16* __restrict__ A) {
  __shared__ alignas(16) __bf16 q_lds[16][64];   // [b][d]
  __shared__ alignas(16) __bf16 w_lds[64][64];   // [n=j][k=d]
  const int tid = threadIdx.x, wid = tid >> 5, lane = tid & 31;
  const int r = lane & 15, hi = lane >> 4;
  const int j0 = blockIdx.x * 64, b0 = blockIdx.y * 16, h = blockIdx.z;

  // qs tile (pure bf16 copy): async DMA, 16 B per thread
  {
    const uint32_t lds_base = (uint32_t)(uintptr_t)(&q_lds[0][0]);
    int bl = tid >> 3, dl8 = (tid & 7) << 3;
    async_b128(lds_base + (uint32_t)tid * 16u,
               qs + (size_t)(b0 + bl) * DMODEL + h * 64 + dl8);
  }
  // Wk tile: load float4 along j (coalesced), transpose-store to [j][d]
#pragma unroll
  for (int it = 0; it < 8; ++it) {
    int g = tid + it * 128;
    int dl = g >> 4, jl4 = (g & 15) << 2;
    float4 v = *(const float4*)(Wk + (size_t)(h * 64 + dl) * DMODEL + j0 + jl4);
    w_lds[jl4 + 0][dl] = (__bf16)v.x;
    w_lds[jl4 + 1][dl] = (__bf16)v.y;
    w_lds[jl4 + 2][dl] = (__bf16)v.z;
    w_lds[jl4 + 3][dl] = (__bf16)v.w;
  }
  async_drain();
  __syncthreads();

  v8f acc;
#pragma unroll
  for (int i = 0; i < 8; ++i) acc[i] = 0.0f;
#pragma unroll
  for (int kb = 0; kb < 64; kb += 32) {
    v16bf a = frag_rowk(&q_lds[0][0], 64, r, kb, hi);
    v16bf b = frag_rowk(&w_lds[0][0], 64, wid * 16 + r, kb, hi);
    acc = __builtin_amdgcn_wmma_f32_16x16x32_bf16(false, a, false, b,
                                                  (short)0, acc, false, false);
  }
#pragma unroll
  for (int i = 0; i < 8; ++i) {
    int bb = b0 + hi * 8 + i;
    int jj = j0 + wid * 16 + r;
    A[((size_t)bb * NHEAD + h) * DMODEL + jj] = (__bf16)acc[i];
  }
}

// ---------------------------------------------------------------------------
// Kernel 3: fused scores -> bias/mask -> online softmax -> ctx (bf16 out).
// One block (8 wave32) per batch. X streamed once from HBM (the 537 MB).
// ---------------------------------------------------------------------------
struct SmemOvl {
  union {
    __bf16 xs[KTILE][DMODEL];   // streamed X tile (bf16)      64 KB
    float  ctx[NHEAD][DMODEL];  // context, written post-loop  64 KB
  };
};

__global__ __launch_bounds__(256) void fattn(
    const float* __restrict__ X, const float* __restrict__ raw,
    const uint8_t* __restrict__ mask, const __bf16* __restrict__ A,
    __bf16* __restrict__ ctx_out,
    const float* __restrict__ lam_d_p, const float* __restrict__ lam_l_p) {
  __shared__ alignas(16) __bf16 A_hk[NHEAD][DMODEL];  // scores B: [n=h][k=j]
  __shared__ alignas(16) SmemOvl ovl;                 // 64 KB
  __shared__ float   part[4][KTILE][NHEAD];           // deterministic partials
  __shared__ float   scomb[KTILE][NHEAD];
  __shared__ alignas(16) __bf16 p_lds[NHEAD][KTILE];  // softmax weights
  __shared__ float   biasv[KTILE];
  __shared__ int     maskv[KTILE];
  __shared__ float   mrun[NHEAD], lrun[NHEAD], resc[NHEAD];

  const int b = blockIdx.x;
  const int tid = threadIdx.x, wid = tid >> 5, lane = tid & 31;
  const int r = lane & 15, hi = lane >> 4;
  const int key = tid >> 3;            // X staging: 8 threads per key row
  const int jb = (tid & 7) << 7;       // 128 contiguous j per thread
  const float lam_d = *lam_d_p, lam_l = *lam_l_p;

  // ---- stage A[b] (pure bf16 copy) with CDNA5 async global->LDS DMA ----
  {
    const __bf16* Ab = A + (size_t)b * NHEAD * DMODEL;
    const uint32_t lds_base = (uint32_t)(uintptr_t)(&A_hk[0][0]);
#pragma unroll
    for (int it = 0; it < 8; ++it) {
      int g = tid + it * 256;  // 2048 groups of 16 bytes
      async_b128(lds_base + (uint32_t)g * 16u, Ab + g * 8);
    }
  }
  if (tid < NHEAD) { mrun[tid] = -3.0e38f; lrun[tid] = 0.0f; }

  v8f acc[8];  // ctx accumulators: wave owns 8 N-tiles (128 j columns)
#pragma unroll
  for (int nt = 0; nt < 8; ++nt)
#pragma unroll
    for (int i = 0; i < 8; ++i) acc[nt][i] = 0.0f;

  async_drain();   // drain this wave's DMA before the barrier
  __syncthreads();

  for (int kt = 0; kt < NKT; ++kt) {
    // ---- stream 32 keys x 1024 dims of X: 2x b128 loads -> 1x b128 store
    {
      const float* xrow =
          X + ((size_t)b * NRES + kt * KTILE + key) * DMODEL + jb;
#pragma unroll
      for (int jj = 0; jj < 128; jj += 8) {
        float4 v0 = *(const float4*)(xrow + jj);
        float4 v1 = *(const float4*)(xrow + jj + 4);
        *(v8bf*)&ovl.xs[key][jb + jj] = pack8(v0, v1);
      }
      // prefetch next tile (global_prefetch_b8): hides HBM latency behind
      // the WMMA + softmax phases of this tile
      if (kt + 1 < NKT) {
        const float* nxt = xrow + KTILE * DMODEL;
        __builtin_prefetch(nxt, 0, 3);
        __builtin_prefetch(nxt + 32, 0, 3);   // +128 B
        __builtin_prefetch(nxt + 64, 0, 3);
        __builtin_prefetch(nxt + 96, 0, 3);
      }
    }
    if (tid < KTILE) {
      const int kk = kt * KTILE + tid;
      const float* rr = raw + ((size_t)b * NRES + kk) * 14;
      biasv[tid] = -(lam_d * rr[2] + lam_l * rr[1]);
      maskv[tid] = (int)mask[(size_t)b * NRES + kk];
    }
    __syncthreads();

    // ---- scores: [32 keys x 1024] @ [1024 x 16 heads], split 2 m x 4 k ----
    {
      const int mt = wid & 1, kc = wid >> 1;
      v8f c;
#pragma unroll
      for (int i = 0; i < 8; ++i) c[i] = 0.0f;
#pragma unroll
      for (int ks = 0; ks < 8; ++ks) {
        const int kb = kc * 256 + ks * 32;
        v16bf a = frag_rowk(&ovl.xs[mt * 16][0], DMODEL, r, kb, hi);
        v16bf bf = frag_rowk(&A_hk[0][0], DMODEL, r, kb, hi);  // row = n = h
        c = __builtin_amdgcn_wmma_f32_16x16x32_bf16(false, a, false, bf,
                                                    (short)0, c, false, false);
      }
#pragma unroll
      for (int i = 0; i < 8; ++i) part[kc][mt * 16 + hi * 8 + i][r] = c[i];
    }
    __syncthreads();

    // ---- online softmax update (one thread per head) ----
    if (tid < NHEAD) {
      const int h = tid;
      const float mo = mrun[h];
      float mn = mo;
      for (int k = 0; k < KTILE; ++k) {
        float s = part[0][k][h] + part[1][k][h] + part[2][k][h] + part[3][k][h];
        s = maskv[k] ? -1.0e9f : (s + biasv[k]);
        scomb[k][h] = s;
        mn = fmaxf(mn, s);
      }
      const float rs = __expf(mo - mn);
      float l = lrun[h] * rs;
      for (int k = 0; k < KTILE; ++k) {
        float p = __expf(scomb[k][h] - mn);
        l += p;
        p_lds[h][k] = (__bf16)p;
      }
      mrun[h] = mn; lrun[h] = l; resc[h] = rs;
    }
    __syncthreads();

    // ---- ctx += p^T @ X_tile : [16h x 32k] @ [32k x 1024j] ----
    {
      float rs[8];
#pragma unroll
      for (int i = 0; i < 8; ++i) rs[i] = resc[hi * 8 + i];
      v16bf pa = frag_rowk(&p_lds[0][0], KTILE, r, 0, hi);  // row = m = h
#pragma unroll
      for (int nt = 0; nt < 8; ++nt) {
#pragma unroll
        for (int i = 0; i < 8; ++i) acc[nt][i] *= rs[i];
        v16bf bf = frag_colgather(&ovl.xs[0][0], DMODEL,
                                  wid * 128 + nt * 16 + r, hi);
        acc[nt] = __builtin_amdgcn_wmma_f32_16x16x32_bf16(
            false, pa, false, bf, (short)0, acc[nt], false, false);
      }
    }
    __syncthreads();  // before next tile overwrites xs
  }

  // ---- normalize, spill ctx to LDS (ovl.xs is dead), stream out as bf16 ----
  {
    float li[8];
#pragma unroll
    for (int i = 0; i < 8; ++i) li[i] = 1.0f / lrun[hi * 8 + i];
#pragma unroll
    for (int nt = 0; nt < 8; ++nt)
#pragma unroll
      for (int i = 0; i < 8; ++i)
        ovl.ctx[hi * 8 + i][wid * 128 + nt * 16 + r] = acc[nt][i] * li[i];
  }
  __syncthreads();
  {
    __bf16* cb = ctx_out + (size_t)b * NHEAD * DMODEL;
    const float* cf = &ovl.ctx[0][0];
#pragma unroll
    for (int it = 0; it < 8; ++it) {
      int g = tid + it * 256;            // 2048 groups of 8
      float4 v0 = *(const float4*)(cf + g * 8);
      float4 v1 = *(const float4*)(cf + g * 8 + 4);
      *(v8bf*)(cb + g * 8) = pack8(v0, v1);
    }
  }
}

// ---------------------------------------------------------------------------
// Kernel 4: tmp[b, h*64+d] = ctx[b,h,:] . Wv[h*64+d,:] + bv   (bf16 out)
// 16 per-head WMMA GEMMs, M=512 (b), N=64 (d), K=1024 (j).
// B = rows of Wv in [n=d][k=j] layout directly (no transpose).
// ---------------------------------------------------------------------------
__global__ __launch_bounds__(128) void vproj(
    const __bf16* __restrict__ ctx, const float* __restrict__ Wv,
    const float* __restrict__ bv, __bf16* __restrict__ tmp) {
  __shared__ alignas(16) __bf16 a_lds[16][64];   // [b][j-chunk]
  __shared__ alignas(16) __bf16 w_lds[64][64];   // [n=d][k=j-chunk]
  const int tid = threadIdx.x, wid = tid >> 5, lane = tid & 31;
  const int r = lane & 15, hi = lane >> 4;
  const int b0 = blockIdx.x * 16, h = blockIdx.y;
  const uint32_t a_base = (uint32_t)(uintptr_t)(&a_lds[0][0]);

  v8f acc;
#pragma unroll
  for (int i = 0; i < 8; ++i) acc[i] = 0.0f;

  for (int ch = 0; ch < 16; ++ch) {
    const int t0 = ch * 64;
    {  // ctx tile (pure bf16 copy): async DMA, 16 B per thread
      int bl = tid >> 3, jl8 = (tid & 7) << 3;
      async_b128(a_base + (uint32_t)tid * 16u,
                 ctx + ((size_t)(b0 + bl) * NHEAD + h) * DMODEL + t0 + jl8);
    }
#pragma unroll
    for (int it = 0; it < 4; ++it) {  // Wv tile: rows are already [n][k]
      int g = tid + it * 128;
      int dl = g >> 3, jl8 = (g & 7) << 3;
      const float* src = Wv + (size_t)(h * 64 + dl) * DMODEL + t0 + jl8;
      *(v8bf*)&w_lds[dl][jl8] =
          pack8(*(const float4*)src, *(const float4*)(src + 4));
    }
    async_drain();
    __syncthreads();
#pragma unroll
    for (int kb = 0; kb < 64; kb += 32) {
      v16bf a = frag_rowk(&a_lds[0][0], 64, r, kb, hi);
      v16bf b = frag_rowk(&w_lds[0][0], 64, wid * 16 + r, kb, hi);
      acc = __builtin_amdgcn_wmma_f32_16x16x32_bf16(false, a, false, b,
                                                    (short)0, acc, false, false);
    }
    __syncthreads();
  }
#pragma unroll
  for (int i = 0; i < 8; ++i) {
    int bb = b0 + hi * 8 + i;
    int dd = h * 64 + wid * 16 + r;
    tmp[(size_t)bb * DMODEL + dd] = (__bf16)(acc[i] + bv[dd]);
  }
}

// ---------------------------------------------------------------------------
// Kernel 5: out[b,c] = tmp[b,:] . Wo[c,:] + bo[c]   (f32 out)
// WMMA GEMM, M=512 (b), N=1024 (c), K=1024 (t).
// ---------------------------------------------------------------------------
__global__ __launch_bounds__(128) void oproj(
    const __bf16* __restrict__ tmp, const float* __restrict__ Wo,
    const float* __restrict__ bo, float* __restrict__ out) {
  __shared__ alignas(16) __bf16 a_lds[16][64];
  __shared__ alignas(16) __bf16 w_lds[64][64];   // [n=c][k=t]
  const int tid = threadIdx.x, wid = tid >> 5, lane = tid & 31;
  const int r = lane & 15, hi = lane >> 4;
  const int c0 = blockIdx.x * 64, b0 = blockIdx.y * 16;
  const uint32_t a_base = (uint32_t)(uintptr_t)(&a_lds[0][0]);

  v8f acc;
#pragma unroll
  for (int i = 0; i < 8; ++i) acc[i] = 0.0f;

  for (int ch = 0; ch < 16; ++ch) {
    const int t0 = ch * 64;
    {  // tmp tile (pure bf16 copy): async DMA, 16 B per thread
      int bl = tid >> 3, tl8 = (tid & 7) << 3;
      async_b128(a_base + (uint32_t)tid * 16u,
                 tmp + (size_t)(b0 + bl) * DMODEL + t0 + tl8);
    }
#pragma unroll
    for (int it = 0; it < 4; ++it) {
      int g = tid + it * 128;
      int cl = g >> 3, tl8 = (g & 7) << 3;
      const float* src = Wo + (size_t)(c0 + cl) * DMODEL + t0 + tl8;
      *(v8bf*)&w_lds[cl][tl8] =
          pack8(*(const float4*)src, *(const float4*)(src + 4));
    }
    async_drain();
    __syncthreads();
#pragma unroll
    for (int kb = 0; kb < 64; kb += 32) {
      v16bf a = frag_rowk(&a_lds[0][0], 64, r, kb, hi);
      v16bf b = frag_rowk(&w_lds[0][0], 64, wid * 16 + r, kb, hi);
      acc = __builtin_amdgcn_wmma_f32_16x16x32_bf16(false, a, false, b,
                                                    (short)0, acc, false, false);
    }
    __syncthreads();
  }
#pragma unroll
  for (int i = 0; i < 8; ++i) {
    int bb = b0 + hi * 8 + i;
    int cc = c0 + wid * 16 + r;
    out[(size_t)bb * DMODEL + cc] = acc[i] + bo[cc];
  }
}

// ---------------------------------------------------------------------------
extern "C" void kernel_launch(void* const* d_in, const int* in_sizes, int n_in,
                              void* d_out, int out_size, void* d_ws,
                              size_t ws_size, hipStream_t stream) {
  (void)in_sizes; (void)n_in; (void)out_size; (void)ws_size;
  const float*   query = (const float*)d_in[0];
  const float*   X     = (const float*)d_in[1];
  const float*   raw   = (const float*)d_in[2];
  const uint8_t* mask  = (const uint8_t*)d_in[3];  // jax bool = 1 byte/elt
  const float*   Wq    = (const float*)d_in[4];
  const float*   bq    = (const float*)d_in[5];
  const float*   Wk    = (const float*)d_in[6];
  // d_in[7] = bk: per-(b,h) constant across keys -> cancels in softmax.
  const float*   Wv    = (const float*)d_in[8];
  const float*   bv    = (const float*)d_in[9];
  const float*   Wo    = (const float*)d_in[10];
  const float*   bo    = (const float*)d_in[11];
  const float*   lamd  = (const float*)d_in[12];
  const float*   laml  = (const float*)d_in[13];
  float* out = (float*)d_out;

  // workspace: qs (1 MB) + A (16 MB) + ctx (16 MB) + tmp (1 MB), all bf16
  __bf16* qs_ws  = reinterpret_cast<__bf16*>(d_ws);
  __bf16* A_ws   = qs_ws + (size_t)NB * DMODEL;
  __bf16* ctx_ws = A_ws + (size_t)NB * NHEAD * DMODEL;
  __bf16* tmp_ws = ctx_ws + (size_t)NB * NHEAD * DMODEL;

  qproj_kernel<<<dim3(DMODEL / 64, NB / 16), 128, 0, stream>>>(query, Wq, bq,
                                                               qs_ws);
  akern<<<dim3(DMODEL / 64, NB / 16, NHEAD), 128, 0, stream>>>(qs_ws, Wk,
                                                               A_ws);
  fattn<<<NB, 256, 0, stream>>>(X, raw, mask, A_ws, ctx_ws, lamd, laml);
  vproj<<<dim3(NB / 16, NHEAD), 128, 0, stream>>>(ctx_ws, Wv, bv, tmp_ws);
  oproj<<<dim3(DMODEL / 64, NB / 16), 128, 0, stream>>>(tmp_ws, Wo, bo, out);
}
